// AsymmetricAttention_43731357007959
// MI455X (gfx1250) — matrix operand order repaired and checked by reading, not compiled, not run on hardware
//
#include <hip/hip_runtime.h>
#include <hip/hip_bf16.h>
#include <stdint.h>

typedef __attribute__((ext_vector_type(16))) __bf16 v16bf;
typedef __attribute__((ext_vector_type(8)))  float  v8f;
typedef __attribute__((ext_vector_type(4)))  unsigned int v4u;
typedef __attribute__((ext_vector_type(4)))  int    v4i;
typedef __attribute__((ext_vector_type(8)))  int    v8i;

#define DEVINL __device__ __forceinline__

#if defined(__has_builtin)
#if __has_builtin(__builtin_amdgcn_tensor_load_to_lds) && \
    __has_builtin(__builtin_amdgcn_s_wait_tensorcnt)
#define HAVE_TDM 1
#endif
#endif

// Problem sizes (B=1)
static constexpr int NX  = 2048;   // x tokens
static constexpr int LY  = 256;    // y tokens
static constexpr int TT  = 2304;   // total tokens
static constexpr int DX  = 3072;   // model dim (x) == H*HD
static constexpr int DY  = 1536;   // model dim (y)
static constexpr int HH  = 24;     // heads
static constexpr int HD  = 128;    // head dim
static constexpr int QKVD = 3 * DX; // 9216

DEVINL int a_koff(int j, int half) {
  // 16-bit A-matrix 16x32 layout: lanes0-15 K=0..7,16..23 ; lanes16-31 K=8..15,24..31
  return (j < 4) ? (half * 8 + 2 * j) : (16 + half * 8 + 2 * (j - 4));
}

DEVINL v8f wmma_bf16(v16bf a, v16bf b, v8f c) {
  return __builtin_amdgcn_wmma_f32_16x16x32_bf16(false, a, false, b, (short)0, c,
                                                 false, false);
}

#if defined(HAVE_TDM)
// ---- Tensor Data Mover descriptor helpers (D#, CDNA5 ISA)
DEVINL v4u tdm_group0(uint32_t lds_byte_addr, const void* gaddr) {
  const uint64_t ga = (uint64_t)(uintptr_t)gaddr;
  v4u g0;
  g0[0] = 1u;                                    // count=1, user descriptor
  g0[1] = lds_byte_addr;                         // lds_addr [63:32]
  g0[2] = (uint32_t)ga;                          // global_addr [95:64]
  g0[3] = (uint32_t)((ga >> 32) & 0x01FFFFFFu)   // global_addr [120:96]
          | (2u << 30);                          // type=2 ("image")
  return g0;
}

// 2-D tile of 16-bit elements: tile_dim0 x tile_dim1 out of dim0 x dim1 tensor.
// pad_int/pad_amt encode LDS destination padding (see D# group1 bit layout).
DEVINL v8i tdm_group1_2d(uint32_t dim0, uint32_t dim1, uint32_t tile0,
                         uint32_t tile1, uint32_t stride0, uint32_t pad_int,
                         uint32_t pad_amt, uint32_t pad_en) {
  v8i g1;
  g1[0] = (int)((1u << 16)            // data_size = 2 bytes
                | (pad_en << 20) | (pad_int << 22) | (pad_amt << 25));
  g1[1] = (int)((dim0 & 0xFFFFu) << 16);                       // tensor_dim0 lo
  g1[2] = (int)(((dim0 >> 16) & 0xFFFFu) | ((dim1 & 0xFFFFu) << 16));
  g1[3] = (int)(((dim1 >> 16) & 0xFFFFu) | ((tile0 & 0xFFFFu) << 16));
  g1[4] = (int)(tile1 & 0xFFFFu);                              // tile_dim1, tile_dim2=0
  g1[5] = (int)stride0;                                        // tensor_dim0_stride lo
  g1[6] = 0;
  g1[7] = 0;
  return g1;
}

DEVINL void tdm_load(v4u g0, v8i g1) {
  v4i z4 = {0, 0, 0, 0};
#if __clang_major__ >= 23
  v8i z8 = {0, 0, 0, 0, 0, 0, 0, 0};
  __builtin_amdgcn_tensor_load_to_lds(g0, g1, z4, z4, z8, 0);
#else
  __builtin_amdgcn_tensor_load_to_lds(g0, g1, z4, z4, 0);
#endif
}
#endif  // HAVE_TDM

// ---------------------------------------------------------------- f32 -> bf16
__global__ void cvt_bf16_kernel(const float* __restrict__ src,
                                __bf16* __restrict__ dst, long n) {
  long stride = (long)gridDim.x * blockDim.x;
  for (long i = (long)blockIdx.x * blockDim.x + threadIdx.x; i < n; i += stride)
    dst[i] = (__bf16)src[i];
}

// ------------------------------------------------- RMS + (1+scale) modulation
__global__ __launch_bounds__(256) void rmsmod_kernel(
    const float* __restrict__ src, const float* __restrict__ scale,
    __bf16* __restrict__ dst, int dim) {
  const int row = blockIdx.x;
  const float* p = src + (size_t)row * dim;
  float ss = 0.f;
  for (int d = threadIdx.x; d < dim; d += 256) { float v = p[d]; ss += v * v; }
  #pragma unroll
  for (int o = 16; o > 0; o >>= 1) ss += __shfl_down(ss, o, 32);
  __shared__ float red[8];
  const int wid = threadIdx.x >> 5, lid = threadIdx.x & 31;
  if (lid == 0) red[wid] = ss;
  __syncthreads();
  float tot = 0.f;
  #pragma unroll
  for (int i = 0; i < 8; ++i) tot += red[i];
  const float inv = rsqrtf(tot / (float)dim + 1e-6f);
  for (int d = threadIdx.x; d < dim; d += 256)
    dst[(size_t)row * dim + d] = (__bf16)(p[d] * inv * (1.0f + scale[d]));
}

// --------------------------------------------- bf16 GEMM: C = A * W^T + bias
// A:[M,K] bf16 rm, W:[Nout,K] bf16 rm, C:[M,Nout] f32. Tiles 128x128x32.
// 256 threads = 8 waves (2x4); wave tile 64x32 = 4x2 WMMA tiles.
// Tiles staged via TDM (tensor_load_to_lds) with double-buffered LDS;
// LDS padding (8 DWORDs per 16-DWORD row) produces the 96B row stride.
__global__ __launch_bounds__(256) void gemm_bf16_nt(
    const __bf16* __restrict__ A, const __bf16* __restrict__ W,
    const float* __restrict__ bias, float* __restrict__ C,
    int M, int Nout, int K) {
  __shared__ __bf16 As[2][128][48];  // stride 96B -> 16B aligned rows
  __shared__ __bf16 Bs[2][128][48];
  const int tid  = threadIdx.x;
  const int wave = tid >> 5, lane = tid & 31;
  const int half = lane >> 4, l16 = lane & 15;
  const int wm = wave >> 2, wn = wave & 3;
  const int mbase = blockIdx.y * 128, nbase = blockIdx.x * 128;
  const int nsteps = K / 32;

  v8f acc[4][2];
  #pragma unroll
  for (int mt = 0; mt < 4; ++mt)
    #pragma unroll
    for (int nt = 0; nt < 2; ++nt)
      #pragma unroll
      for (int e = 0; e < 8; ++e) acc[mt][nt][e] = 0.0f;

#if defined(HAVE_TDM)
  const bool issuer = (tid < 32);          // wave 0 programs the TDM
  const v8i g1A = tdm_group1_2d(K, M,    32, 128, K, /*pad_int 16dw*/3,
                                /*pad_amt 8dw*/7, 1);
  const v8i g1B = tdm_group1_2d(K, Nout, 32, 128, K, 3, 7, 1);
  if (issuer) {
    tdm_load(tdm_group0((uint32_t)(uintptr_t)&As[0][0][0],
                        A + (size_t)mbase * K), g1A);
    tdm_load(tdm_group0((uint32_t)(uintptr_t)&Bs[0][0][0],
                        W + (size_t)nbase * K), g1B);
  }
  for (int s = 0; s < nsteps; ++s) {
    const int buf = s & 1;
    if (issuer) {
      if (s + 1 < nsteps) {   // prefetch next tile pair into the other buffer
        tdm_load(tdm_group0((uint32_t)(uintptr_t)&As[buf ^ 1][0][0],
                            A + (size_t)mbase * K + (s + 1) * 32), g1A);
        tdm_load(tdm_group0((uint32_t)(uintptr_t)&Bs[buf ^ 1][0][0],
                            W + (size_t)nbase * K + (s + 1) * 32), g1B);
        __builtin_amdgcn_s_wait_tensorcnt(2);   // oldest pair (step s) done
      } else {
        __builtin_amdgcn_s_wait_tensorcnt(0);
      }
    }
    __syncthreads();
    union { v16bf v; uint32_t u[8]; } af[4], bfr[2];
    #pragma unroll
    for (int mt = 0; mt < 4; ++mt) {
      const __bf16* rp = &As[buf][wm * 64 + mt * 16 + l16][0];
      #pragma unroll
      for (int j = 0; j < 8; ++j) af[mt].u[j] = *(const uint32_t*)(rp + a_koff(j, half));
    }
    #pragma unroll
    for (int nt = 0; nt < 2; ++nt) {
      const __bf16* rp = &Bs[buf][wn * 32 + nt * 16 + l16][0];
      #pragma unroll
      for (int j = 0; j < 8; ++j) bfr[nt].u[j] = *(const uint32_t*)(rp + a_koff(j, half));
    }
    #pragma unroll
    for (int mt = 0; mt < 4; ++mt)
      #pragma unroll
      for (int nt = 0; nt < 2; ++nt)
        acc[mt][nt] = wmma_bf16(af[mt].v, bfr[nt].v, acc[mt][nt]);
    __syncthreads();   // all reads of buf done before TDM overwrites it
  }
#else
  for (int k0 = 0; k0 < K; k0 += 32) {
    __syncthreads();
    #pragma unroll
    for (int it = 0; it < 2; ++it) {   // 128x32 bf16 = 512 x 16B chunks
      int c  = tid + it * 256;
      int r  = c >> 2, kc = (c & 3) << 3;
      *(float4*)&As[0][r][kc] = *(const float4*)(A + (size_t)(mbase + r) * K + k0 + kc);
      *(float4*)&Bs[0][r][kc] = *(const float4*)(W + (size_t)(nbase + r) * K + k0 + kc);
    }
    __syncthreads();
    union { v16bf v; uint32_t u[8]; } af[4], bfr[2];
    #pragma unroll
    for (int mt = 0; mt < 4; ++mt) {
      const __bf16* rp = &As[0][wm * 64 + mt * 16 + l16][0];
      #pragma unroll
      for (int j = 0; j < 8; ++j) af[mt].u[j] = *(const uint32_t*)(rp + a_koff(j, half));
    }
    #pragma unroll
    for (int nt = 0; nt < 2; ++nt) {
      const __bf16* rp = &Bs[0][wn * 32 + nt * 16 + l16][0];
      #pragma unroll
      for (int j = 0; j < 8; ++j) bfr[nt].u[j] = *(const uint32_t*)(rp + a_koff(j, half));
    }
    #pragma unroll
    for (int mt = 0; mt < 4; ++mt)
      #pragma unroll
      for (int nt = 0; nt < 2; ++nt)
        acc[mt][nt] = wmma_bf16(af[mt].v, bfr[nt].v, acc[mt][nt]);
  }
#endif

  #pragma unroll
  for (int mt = 0; mt < 4; ++mt)
    #pragma unroll
    for (int nt = 0; nt < 2; ++nt) {
      const int gm = mbase + wm * 64 + mt * 16 + half * 8;
      const int gn = nbase + wn * 32 + nt * 16 + l16;
      const float bv = bias[gn];
      #pragma unroll
      for (int j = 0; j < 8; ++j)
        C[(size_t)(gm + j) * Nout + gn] = acc[mt][nt][j] + bv;
    }
}

// --------------------------- per-head RMS (+RoPE for x); scatter q/k/v bf16
__global__ __launch_bounds__(128) void qkv_post_kernel(
    const float* __restrict__ qkv,  // [tokens, 9216] bias already added
    const float* __restrict__ wq, const float* __restrict__ wk,
    const float* __restrict__ cosb, const float* __restrict__ sinb, // [tok,H,64]
    __bf16* __restrict__ Qo, __bf16* __restrict__ Ko, __bf16* __restrict__ Vo,
    int use_rope, int t_offset) {
  const int i = blockIdx.x, h = blockIdx.y, d = threadIdx.x;
  const float* base = qkv + (size_t)i * QKVD + h * HD;
  float qv = base[d], kv = base[DX + d], vv = base[2 * DX + d];
  __shared__ float sq[128], sk[128], redq[4], redk[4];
  float s1 = qv * qv, s2 = kv * kv;
  #pragma unroll
  for (int o = 16; o > 0; o >>= 1) {
    s1 += __shfl_down(s1, o, 32);
    s2 += __shfl_down(s2, o, 32);
  }
  if ((d & 31) == 0) { redq[d >> 5] = s1; redk[d >> 5] = s2; }
  __syncthreads();
  const float ssq = redq[0] + redq[1] + redq[2] + redq[3];
  const float ssk = redk[0] + redk[1] + redk[2] + redk[3];
  float qn = qv * rsqrtf(ssq * (1.0f / 128.0f) + 1e-5f) * wq[d];
  float kn = kv * rsqrtf(ssk * (1.0f / 128.0f) + 1e-5f) * wk[d];
  if (use_rope) {
    sq[d] = qn; sk[d] = kn;
    __syncthreads();
    const int m = d >> 1;
    const float c = cosb[((size_t)i * HH + h) * 64 + m];
    const float s = sinb[((size_t)i * HH + h) * 64 + m];
    const float qe = sq[2 * m], qo2 = sq[2 * m + 1];
    const float ke = sk[2 * m], ko2 = sk[2 * m + 1];
    qn = (d & 1) ? (qe * s + qo2 * c) : (qe * c - qo2 * s);
    kn = (d & 1) ? (ke * s + ko2 * c) : (ke * c - ko2 * s);
  }
  const size_t oidx = (size_t)(t_offset + i) * DX + h * HD + d;
  Qo[oidx] = (__bf16)qn;
  Ko[oidx] = (__bf16)kn;
  Vo[oidx] = (__bf16)vv;
}

// ------------------------------------------------- flash attention (bf16 WMMA)
// grid (TT/128, H); 256 threads = 8 waves, each wave owns 16 q rows.
// K chunks streamed by TDM (double-buffered LDS), V chunks pipelined through
// VGPRs one chunk ahead, so both overlap with the WMMA work of each chunk.
__global__ __launch_bounds__(256) void attn_kernel(
    const __bf16* __restrict__ Q, const __bf16* __restrict__ Kb,
    const __bf16* __restrict__ Vb, const int* __restrict__ seqlens,
    __bf16* __restrict__ O) {
#if defined(HAVE_TDM)
  __shared__ __bf16 Ks[2][128][136];   // [key][d], stride 272B (16B aligned)
#else
  __shared__ __bf16 Ks[1][128][136];
#endif
  __shared__ __bf16 Vt[128][136];      // transposed [d][key]
  __shared__ __bf16 Ps[8][16][136];    // per-wave P tile (A-layout staging)
  const int tid  = threadIdx.x, wave = tid >> 5, lane = tid & 31;
  const int half = lane >> 4, l16 = lane & 15;
  const int h = blockIdx.y;
  const int qbase = blockIdx.x * 128 + wave * 16;
  const int Tvalid = NX + seqlens[0];
  const float scale = 0.08838834764831845f;   // HD^-0.5
  const float LOG2E = 1.4426950408889634f;
  constexpr int NC = TT / 128;

  union Frag { v16bf v; uint32_t u[8]; };
  Frag qf[4];
  {
    const __bf16* qrow = Q + (size_t)(qbase + l16) * DX + h * HD;
    #pragma unroll
    for (int kd = 0; kd < 4; ++kd)
      #pragma unroll
      for (int j = 0; j < 8; ++j)
        qf[kd].u[j] = *(const uint32_t*)(qrow + kd * 32 + a_koff(j, half));
  }

  // chunk-ahead register pipeline (V always; K too when no TDM)
  float4 vreg[8];
#if !defined(HAVE_TDM)
  float4 kreg[8];
#endif
  auto gload = [&](int kb) {
    #pragma unroll
    for (int it = 0; it < 8; ++it) {
      int c = tid + it * 256;
      int r = c >> 4, dc = (c & 15) << 3;
      vreg[it] = *(const float4*)(Vb + (size_t)(kb + r) * DX + h * HD + dc);
#if !defined(HAVE_TDM)
      kreg[it] = *(const float4*)(Kb + (size_t)(kb + r) * DX + h * HD + dc);
#endif
    }
  };

#if defined(HAVE_TDM)
  const bool issuer = (tid < 32);
  // 128x128 bf16 tile; LDS pad 4 DWORDs per 64-DWORD row -> 272B row stride
  const v8i g1K = tdm_group1_2d(DX, TT, 128, 128, DX, /*pad_int 64dw*/5,
                                /*pad_amt 4dw*/3, 1);
  if (issuer)
    tdm_load(tdm_group0((uint32_t)(uintptr_t)&Ks[0][0][0], Kb + h * HD), g1K);
#endif
  gload(0);

  v8f oacc[8];
  #pragma unroll
  for (int nt = 0; nt < 8; ++nt)
    #pragma unroll
    for (int e = 0; e < 8; ++e) oacc[nt][e] = 0.0f;
  float mstate[8], lstate[8];
  #pragma unroll
  for (int j = 0; j < 8; ++j) { mstate[j] = -3.0e38f; lstate[j] = 0.0f; }

  for (int kc = 0; kc < NC; ++kc) {
    const int kbase = kc * 128;
    __syncthreads();   // A: all waves done with previous chunk's LDS
#if defined(HAVE_TDM)
    const int kbuf = kc & 1;
    if (issuer && kc + 1 < NC)   // DMA next K chunk while we compute this one
      tdm_load(tdm_group0((uint32_t)(uintptr_t)&Ks[kbuf ^ 1][0][0],
                          Kb + (size_t)(kbase + 128) * DX + h * HD), g1K);
#else
    const int kbuf = 0;
#endif
    #pragma unroll
    for (int it = 0; it < 8; ++it) {   // drain V regs (chunk kc) into Vt^T
      int c = tid + it * 256;
      int r = c >> 4, dc = (c & 15) << 3;
      const __bf16* vp = (const __bf16*)&vreg[it];
      #pragma unroll
      for (int e = 0; e < 8; ++e) Vt[dc + e][r] = vp[e];
#if !defined(HAVE_TDM)
      *(float4*)&Ks[0][r][dc] = kreg[it];
#endif
    }
#if defined(HAVE_TDM)
    if (issuer) {
      if (kc + 1 < NC) {
        __builtin_amdgcn_s_wait_tensorcnt(1);   // chunk kc landed, kc+1 in flight
      } else {
        __builtin_amdgcn_s_wait_tensorcnt(0);   // last chunk landed
      }
    }
#endif
    if (kc + 1 < NC) gload(kbase + 128);  // next V (and K) loads fly over compute
    __syncthreads();   // B: Ks[kbuf] + Vt visible to all waves

    // S = Q K^T  (16 x 128 per wave)
    v8f sacc[8];
    #pragma unroll
    for (int nt = 0; nt < 8; ++nt) {
      #pragma unroll
      for (int e = 0; e < 8; ++e) sacc[nt][e] = 0.0f;
      #pragma unroll
      for (int kd = 0; kd < 4; ++kd) {
        Frag kf;
        #pragma unroll
        for (int j = 0; j < 8; ++j)
          kf.u[j] = *(const uint32_t*)(&Ks[kbuf][nt * 16 + l16][kd * 32 + a_koff(j, half)]);
        sacc[nt] = wmma_bf16(qf[kd].v, kf.v, sacc[nt]);
      }
    }
    // scale + key mask + online softmax
    float rmax[8];
    #pragma unroll
    for (int j = 0; j < 8; ++j) rmax[j] = -3.0e38f;
    #pragma unroll
    for (int nt = 0; nt < 8; ++nt) {
      const bool kvalid = (kbase + nt * 16 + l16) < Tvalid;
      #pragma unroll
      for (int j = 0; j < 8; ++j) {
        float sv = kvalid ? sacc[nt][j] * scale : -1.0e30f;
        sacc[nt][j] = sv;
        rmax[j] = fmaxf(rmax[j], sv);
      }
    }
    #pragma unroll
    for (int j = 0; j < 8; ++j) {
      float v = rmax[j];
      #pragma unroll
      for (int o = 1; o < 16; o <<= 1) v = fmaxf(v, __shfl_xor(v, o, 32));
      const float mnew = fmaxf(mstate[j], v);
      rmax[j] = exp2f((mstate[j] - mnew) * LOG2E);  // correction factor
      mstate[j] = mnew;
    }
    float rsum[8];
    #pragma unroll
    for (int j = 0; j < 8; ++j) rsum[j] = 0.0f;
    #pragma unroll
    for (int nt = 0; nt < 8; ++nt)
      #pragma unroll
      for (int j = 0; j < 8; ++j) {
        float p = exp2f((sacc[nt][j] - mstate[j]) * LOG2E);
        rsum[j] += p;
        Ps[wave][half * 8 + j][nt * 16 + l16] = (__bf16)p;
      }
    #pragma unroll
    for (int j = 0; j < 8; ++j) {
      float v = rsum[j];
      #pragma unroll
      for (int o = 1; o < 16; o <<= 1) v += __shfl_xor(v, o, 32);
      lstate[j] = lstate[j] * rmax[j] + v;
      #pragma unroll
      for (int nt = 0; nt < 8; ++nt) oacc[nt][j] *= rmax[j];
    }
    // O += P V   (P 16x128 from LDS, V^T in LDS)
    Frag pf[4];
    #pragma unroll
    for (int kd = 0; kd < 4; ++kd)
      #pragma unroll
      for (int j = 0; j < 8; ++j)
        pf[kd].u[j] = *(const uint32_t*)(&Ps[wave][l16][kd * 32 + a_koff(j, half)]);
    #pragma unroll
    for (int nt = 0; nt < 8; ++nt)
      #pragma unroll
      for (int kd = 0; kd < 4; ++kd) {
        Frag vf;
        #pragma unroll
        for (int j = 0; j < 8; ++j)
          vf.u[j] = *(const uint32_t*)(&Vt[nt * 16 + l16][kd * 32 + a_koff(j, half)]);
        oacc[nt] = wmma_bf16(pf[kd].v, vf.v, oacc[nt]);
      }
  }
  // epilogue: normalize, zero invalid query rows, store bf16
  #pragma unroll
  for (int j = 0; j < 8; ++j) {
    const int t = qbase + half * 8 + j;
    const bool qvalid = t < Tvalid;        // t < NX always valid
    const float inv = 1.0f / lstate[j];
    #pragma unroll
    for (int nt = 0; nt < 8; ++nt) {
      const float ov = qvalid ? oacc[nt][j] * inv : 0.0f;
      O[(size_t)t * DX + h * HD + nt * 16 + l16] = (__bf16)ov;
    }
  }
}

// ============================================================== host launcher
extern "C" void kernel_launch(void* const* d_in, const int* in_sizes, int n_in,
                              void* d_out, int out_size, void* d_ws, size_t ws_size,
                              hipStream_t stream) {
  const float* x        = (const float*)d_in[0];
  const float* y        = (const float*)d_in[1];
  const float* scale_x  = (const float*)d_in[2];
  const float* scale_y  = (const float*)d_in[3];
  const float* rope_cos = (const float*)d_in[4];
  const float* rope_sin = (const float*)d_in[5];
  const int*   seqlens  = (const int*)d_in[6];
  const float* W_qkv_x  = (const float*)d_in[7];
  const float* b_qkv_x  = (const float*)d_in[8];
  const float* W_qkv_y  = (const float*)d_in[9];
  const float* b_qkv_y  = (const float*)d_in[10];
  const float* w_qnorm_x = (const float*)d_in[11];
  const float* w_knorm_x = (const float*)d_in[12];
  const float* w_qnorm_y = (const float*)d_in[13];
  const float* w_knorm_y = (const float*)d_in[14];
  const float* W_proj_x  = (const float*)d_in[15];
  const float* b_proj_x  = (const float*)d_in[16];
  const float* W_proj_y  = (const float*)d_in[17];
  const float* b_proj_y  = (const float*)d_in[18];

  char* ws = (char*)d_ws;
  size_t off = 0;
  auto alloc = [&](size_t bytes) -> void* {
    void* p = ws + off;
    off += (bytes + 255) & ~(size_t)255;
    return p;
  };
  __bf16* Wqkvx_bf  = (__bf16*)alloc((size_t)QKVD * DX * 2);
  __bf16* Wqkvy_bf  = (__bf16*)alloc((size_t)QKVD * DY * 2);
  __bf16* Wprojx_bf = (__bf16*)alloc((size_t)DX * DX * 2);
  __bf16* Wprojy_bf = (__bf16*)alloc((size_t)DY * DX * 2);
  __bf16* xm_bf     = (__bf16*)alloc((size_t)NX * DX * 2);
  __bf16* ym_bf     = (__bf16*)alloc((size_t)LY * DY * 2);
  float*  qkvx_f    = (float*)alloc((size_t)NX * QKVD * 4);
  float*  qkvy_f    = (float*)alloc((size_t)LY * QKVD * 4);
  __bf16* q_bf      = (__bf16*)alloc((size_t)TT * DX * 2);
  __bf16* k_bf      = (__bf16*)alloc((size_t)TT * DX * 2);
  __bf16* v_bf      = (__bf16*)alloc((size_t)TT * DX * 2);
  __bf16* attn_bf   = (__bf16*)alloc((size_t)TT * DX * 2);

  // 1) weights f32 -> bf16
  cvt_bf16_kernel<<<4096, 256, 0, stream>>>(W_qkv_x, Wqkvx_bf, (long)QKVD * DX);
  cvt_bf16_kernel<<<4096, 256, 0, stream>>>(W_qkv_y, Wqkvy_bf, (long)QKVD * DY);
  cvt_bf16_kernel<<<2048, 256, 0, stream>>>(W_proj_x, Wprojx_bf, (long)DX * DX);
  cvt_bf16_kernel<<<2048, 256, 0, stream>>>(W_proj_y, Wprojy_bf, (long)DY * DX);

  // 2) RMS + modulation
  rmsmod_kernel<<<NX, 256, 0, stream>>>(x, scale_x, xm_bf, DX);
  rmsmod_kernel<<<LY, 256, 0, stream>>>(y, scale_y, ym_bf, DY);

  // 3) QKV GEMMs (bias fused)
  gemm_bf16_nt<<<dim3(QKVD / 128, NX / 128), 256, 0, stream>>>(
      xm_bf, Wqkvx_bf, b_qkv_x, qkvx_f, NX, QKVD, DX);
  gemm_bf16_nt<<<dim3(QKVD / 128, LY / 128), 256, 0, stream>>>(
      ym_bf, Wqkvy_bf, b_qkv_y, qkvy_f, LY, QKVD, DY);

  // 4) per-head norms (+RoPE for x), scatter to [T, H*HD] bf16
  qkv_post_kernel<<<dim3(NX, HH), 128, 0, stream>>>(
      qkvx_f, w_qnorm_x, w_knorm_x, rope_cos, rope_sin, q_bf, k_bf, v_bf, 1, 0);
  qkv_post_kernel<<<dim3(LY, HH), 128, 0, stream>>>(
      qkvy_f, w_qnorm_y, w_knorm_y, nullptr, nullptr, q_bf, k_bf, v_bf, 0, NX);

  // 5) attention
  attn_kernel<<<dim3(TT / 128, HH), 256, 0, stream>>>(q_bf, k_bf, v_bf, seqlens,
                                                      attn_bf);

  // 6) projections straight into d_out (f32)
  float* x_out = (float*)d_out;
  float* y_out = x_out + (size_t)NX * DX;
  gemm_bf16_nt<<<dim3(DX / 128, NX / 128), 256, 0, stream>>>(
      attn_bf, Wprojx_bf, b_proj_x, x_out, NX, DX, DX);
  gemm_bf16_nt<<<dim3(DY / 128, LY / 128), 256, 0, stream>>>(
      attn_bf + (size_t)NX * DX, Wprojy_bf, b_proj_y, y_out, LY, DY, DX);
}
